// TransLayer_84928683311329
// MI455X (gfx1250) — matrix operand order repaired and checked by reading, not compiled, not run on hardware
//
#include <hip/hip_runtime.h>
#include <hip/hip_bf16.h>
#include <stdint.h>

// ---------------- problem constants ----------------
constexpr int kDim   = 512;
constexpr int kHeads = 8;
constexpr int kDh    = 64;
constexpr int kM     = 256;           // landmarks
constexpr int kB     = 4;
constexpr int kN     = 8192;
constexpr int kBH    = kB * kHeads;   // 32
constexpr int kNT    = kB * kN;       // 32768 tokens
constexpr int kL     = kN / kM;       // 32 tokens per landmark group
constexpr int kSplit = 8;             // flash splits for attn3@v
constexpr float kLnEps = 1e-5f;

// padded LDS strides (all multiples of 8 elements -> b128-aligned fragments)
constexpr int kSA   = 40;   // bf16 A-tile stride (32 used)
constexpr int kSB   = 48;   // bf16 B-tile stride (32 used)
constexpr int kSAf  = 36;   // f32 A/B tile stride (32 used)
constexpr int kSK   = 72;   // attn3v k-chunk stride (64 used)
constexpr int kSV   = 136;  // attn3v vT / P stride (128 used)
constexpr int kSP   = 264;  // attn1 P stride (256 used)

typedef unsigned short bf16_t;
typedef __attribute__((ext_vector_type(16))) __bf16 v16bf;
typedef __attribute__((ext_vector_type(8)))  float  v8f;
typedef __attribute__((ext_vector_type(2)))  float  v2f;

union BFrag { v16bf v; bf16_t s[16]; unsigned int u[8]; };
union FAcc  { v8f v; float f[8]; };

__device__ __forceinline__ bf16_t f2bf(float f){
  union { float f; uint32_t u; } c; c.f = f;
  uint32_t u = c.u;
  return (bf16_t)((u + 0x7fffu + ((u >> 16) & 1u)) >> 16);   // RNE
}
__device__ __forceinline__ float bf2f(bf16_t b){
  union { uint32_t u; float f; } c; c.u = ((uint32_t)b) << 16;
  return c.f;
}

__device__ __forceinline__ v8f wmma_bf16(v16bf a, v16bf b, v8f c){
  return __builtin_amdgcn_wmma_f32_16x16x32_bf16(false, a, false, b, (short)0, c, false, false);
}
__device__ __forceinline__ v8f wmma_f32(v2f a, v2f b, v8f c){
  return __builtin_amdgcn_wmma_f32_16x16x4_f32(false, a, false, b, (short)0, c, false, false);
}

// A fragment, 16x32 bf16 step: lane&15 = row, lane>>4 picks K {0..7,16..23}/{8..15,24..31}
__device__ __forceinline__ void ldA(BFrag& fr, const bf16_t* base, int stride, int row, int kof, int kh){
  const bf16_t* p = base + (size_t)row * stride + kof + kh * 8;
  *(uint4*)&fr.u[0] = *(const uint4*)p;
  *(uint4*)&fr.u[4] = *(const uint4*)(p + 16);
}
// B fragment, 32x16 bf16 step from K-contiguous (transposed) storage:
// lane&15 = col, lane>>4 picks K {0..7,16..23}/{8..15,24..31}
__device__ __forceinline__ void ldB(BFrag& fr, const bf16_t* baseT, int stride, int col, int kof, int kh){
  const bf16_t* p = baseT + (size_t)col * stride + kof + kh * 8;
  *(uint4*)&fr.u[0] = *(const uint4*)p;
  *(uint4*)&fr.u[4] = *(const uint4*)(p + 16);
}

// row reductions across the 16 lanes of a wave half (C-fragment rows)
__device__ __forceinline__ float hredMax(float v){
  #pragma unroll
  for (int m = 1; m < 16; m <<= 1) v = fmaxf(v, __shfl_xor(v, m, 32));
  return v;
}
__device__ __forceinline__ float hredSum(float v){
  #pragma unroll
  for (int m = 1; m < 16; m <<= 1) v += __shfl_xor(v, m, 32);
  return v;
}

// ========== 0) one-shot: transpose+convert weights to bf16, K-contiguous ==========
// wT[c*K + k] = bf16(w[k*C + c])
__global__ void convert_wT(const float* __restrict__ w, bf16_t* __restrict__ wT, int K, int C)
{
  int t = blockIdx.x * 256 + threadIdx.x;
  int c = t >> 9, k = t & 511;                  // K == 512 for both weights
  wT[t] = f2bf(w[(size_t)k * C + c]);
}

// ================= 1) LayerNorm -> bf16 (one token per wave) =================
__global__ void ln_kernel(const float* __restrict__ x, const float* __restrict__ g,
                          const float* __restrict__ bta, bf16_t* __restrict__ h)
{
  int wave = threadIdx.x >> 5, lane = threadIdx.x & 31;
  int token = blockIdx.x * 8 + wave;
  const float* row = x + (size_t)token * kDim;
  float vals[16], s = 0.f;
  #pragma unroll
  for (int i = 0; i < 16; i++){ vals[i] = row[lane + 32*i]; s += vals[i]; }
  #pragma unroll
  for (int m = 16; m >= 1; m >>= 1) s += __shfl_xor(s, m, 32);
  float mu = s * (1.0f / kDim), vs = 0.f;
  #pragma unroll
  for (int i = 0; i < 16; i++){ float d = vals[i] - mu; vs += d * d; }
  #pragma unroll
  for (int m = 16; m >= 1; m >>= 1) vs += __shfl_xor(vs, m, 32);
  float rstd = rsqrtf(vs * (1.0f / kDim) + kLnEps);
  bf16_t* orow = h + (size_t)token * kDim;
  #pragma unroll
  for (int i = 0; i < 16; i++){
    int c = lane + 32*i;
    orow[c] = f2bf((vals[i] - mu) * rstd * g[c] + bta[c]);
  }
}

// ================= 2) QKV GEMM (bf16 WMMA), 64x128 block tile =================
__global__ void qkv_gemm(const bf16_t* __restrict__ h, const bf16_t* __restrict__ wT,
                         bf16_t* __restrict__ qb, bf16_t* __restrict__ kb, bf16_t* __restrict__ vb)
{
  int rb0 = blockIdx.x * 64, cb0 = blockIdx.y * 128;
  int w = threadIdx.x >> 5, lane = threadIdx.x & 31;
  int lr = lane & 15, kh = lane >> 4;
  int wm = w >> 1, wn = w & 1;
  __shared__ __align__(16) bf16_t As[64 * kSA];
  __shared__ __align__(16) bf16_t Bt[128 * kSB];
  FAcc acc[4];
  #pragma unroll
  for (int jt = 0; jt < 4; jt++) for (int i = 0; i < 8; i++) acc[jt].f[i] = 0.f;
  for (int k0 = 0; k0 < kDim; k0 += 32){
    { int r = threadIdx.x >> 2, c8 = (threadIdx.x & 3) * 8;
      *(uint4*)&As[r*kSA + c8] = *(const uint4*)(h + (size_t)(rb0 + r) * kDim + k0 + c8); }
    #pragma unroll
    for (int g = 0; g < 2; g++){
      int gi = threadIdx.x * 2 + g;                 // 512 8-elem groups: 128 cols x 4
      int col = gi >> 2, k8 = (gi & 3) * 8;
      *(uint4*)&Bt[col*kSB + k8] =
          *(const uint4*)(wT + (size_t)(cb0 + col) * kDim + k0 + k8);
    }
    __syncthreads();
    BFrag a; ldA(a, As, kSA, wm*16 + lr, 0, kh);
    #pragma unroll
    for (int jt = 0; jt < 4; jt++){
      BFrag b; ldB(b, Bt, kSB, wn*64 + jt*16 + lr, 0, kh);
      acc[jt].v = wmma_bf16(a.v, b.v, acc[jt].v);
    }
    __syncthreads();
  }
  // block-uniform section select (128-col tile never crosses q/k/v boundary)
  int sec = cb0 >> 9;
  bf16_t* dst = (sec == 0) ? qb : ((sec == 1) ? kb : vb);
  float scl = (sec == 0) ? 0.125f : 1.0f;          // q * dh^-0.5
  #pragma unroll
  for (int jt = 0; jt < 4; jt++)
    for (int i = 0; i < 8; i++){
      int row = rb0 + wm*16 + i + 8*kh;
      int col = cb0 + wn*64 + jt*16 + lr;
      int hh = (col >> 6) & 7, d = col & 63;
      int b = row >> 13, n = row & (kN - 1);
      size_t o = (((size_t)b * kHeads + hh) * kN + n) * kDh + d;
      dst[o] = f2bf(acc[jt].f[i] * scl);
    }
}

// ================= 3) landmark means =================
__global__ void landmark_kernel(const bf16_t* __restrict__ q, const bf16_t* __restrict__ k,
                                bf16_t* __restrict__ ql, bf16_t* __restrict__ kl)
{
  int m = blockIdx.x & (kM - 1), bh = blockIdx.x >> 8, d = threadIdx.x;
  size_t base = ((size_t)bh * kN + m * kL) * kDh + d;
  float sq = 0.f, sk = 0.f;
  for (int t = 0; t < kL; t++){ sq += bf2f(q[base + t*kDh]); sk += bf2f(k[base + t*kDh]); }
  size_t o = ((size_t)bh * kM + m) * kDh + d;
  ql[o] = f2bf(sq * (1.f / kL)); kl[o] = f2bf(sk * (1.f / kL));
}

// ================= 4) sim2 = qland@kland^T, row softmax (per-wave 16 rows) =================
__global__ void sim2_softmax(const bf16_t* __restrict__ ql, const bf16_t* __restrict__ kl,
                             float* __restrict__ attn2)
{
  int bh = blockIdx.x >> 1;
  int w = threadIdx.x >> 5, lane = threadIdx.x & 31;
  int lr = lane & 15, kh = lane >> 4;
  int rbase = (blockIdx.x & 1) * 128 + w * 16;
  const bf16_t* qbb = ql + (size_t)bh * kM * kDh;
  const bf16_t* kbb = kl + (size_t)bh * kM * kDh;
  BFrag a0, a1;
  ldA(a0, qbb, kDh, rbase + lr, 0, kh);
  ldA(a1, qbb, kDh, rbase + lr, 32, kh);
  FAcc S[16];
  for (int jt = 0; jt < 16; jt++){
    for (int i = 0; i < 8; i++) S[jt].f[i] = 0.f;
    BFrag b0, b1;
    ldB(b0, kbb, kDh, jt*16 + lr, 0, kh);
    ldB(b1, kbb, kDh, jt*16 + lr, 32, kh);
    S[jt].v = wmma_bf16(a0.v, b0.v, S[jt].v);
    S[jt].v = wmma_bf16(a1.v, b1.v, S[jt].v);
  }
  float* outb = attn2 + (size_t)bh * kM * kM;
  #pragma unroll
  for (int i = 0; i < 8; i++){
    float mx = -1e30f;
    for (int jt = 0; jt < 16; jt++) mx = fmaxf(mx, S[jt].f[i]);
    mx = hredMax(mx);
    float sum = 0.f;
    for (int jt = 0; jt < 16; jt++){ float e = __expf(S[jt].f[i] - mx); S[jt].f[i] = e; sum += e; }
    sum = hredSum(sum);
    float inv = 1.f / sum;
    int row = rbase + i + 8*kh;
    for (int jt = 0; jt < 16; jt++)
      outb[(size_t)row * kM + jt*16 + lr] = S[jt].f[i] * inv;
  }
}

// ================= 5) pinv init scale: global max of column sums =================
__global__ void init_scalar(float* p){ if (threadIdx.x == 0) p[0] = 0.f; }

__global__ void colsum_max(const float* __restrict__ attn2, float* __restrict__ pmax)
{
  int bh = blockIdx.x, col = threadIdx.x;
  const float* Mx = attn2 + (size_t)bh * kM * kM;
  float s = 0.f;
  for (int r = 0; r < kM; r++) s += Mx[r * kM + col];
  __shared__ float red[256];
  red[col] = s; __syncthreads();
  for (int st = 128; st > 0; st >>= 1){
    if (col < st) red[col] = fmaxf(red[col], red[col + st]);
    __syncthreads();
  }
  if (col == 0) atomicMax((int*)pmax, __float_as_int(red[0]));   // sums are positive
}

// z = attn2^T / (maxrow * maxcol); softmax rows sum to 1 -> maxrow == 1
__global__ void pinv_init(const float* __restrict__ attn2, const float* __restrict__ pmax,
                          float* __restrict__ z)
{
  int t = blockIdx.x * 512 + threadIdx.x;       // 32*256*256 = 2^21 elems
  float inv = 1.0f / pmax[0];
  int bh = t >> 16, r = (t >> 8) & 255, c = t & 255;
  z[t] = attn2[(size_t)bh * 65536 + (size_t)c * kM + r] * inv;
}

// ================= 6) batched 256^3 f32 GEMM (V_WMMA_F32_16X16X4_F32) =================
// C = coefI*I + coefAdd*Add + coefAB*(A@B)
__global__ void gemm256(const float* __restrict__ A, const float* __restrict__ Bm,
                        const float* __restrict__ Add, float* __restrict__ C,
                        float coefI, float coefAdd, float coefAB)
{
  int bh = blockIdx.z;
  int rb0 = blockIdx.x * 64, cb0 = blockIdx.y * 128;
  const float* Ab = A + (size_t)bh * 65536;
  const float* Bb = Bm + (size_t)bh * 65536;
  float* Cb = C + (size_t)bh * 65536;
  int w = threadIdx.x >> 5, lane = threadIdx.x & 31;
  int lr = lane & 15, k2 = (lane >> 4) * 2;
  int wm = w >> 1, wn = w & 1;
  __shared__ __align__(16) float As[64 * kSAf];
  __shared__ __align__(16) float Bt[128 * kSAf];
  FAcc acc[4];
  #pragma unroll
  for (int jt = 0; jt < 4; jt++) for (int i = 0; i < 8; i++) acc[jt].f[i] = 0.f;
  for (int k0 = 0; k0 < kM; k0 += 32){
    { int r = threadIdx.x >> 2, c8 = (threadIdx.x & 3) * 8;
      const float* src = Ab + (size_t)(rb0 + r) * kM + k0 + c8;
      *(float4*)&As[r*kSAf + c8]     = *(const float4*)src;
      *(float4*)&As[r*kSAf + c8 + 4] = *(const float4*)(src + 4); }
    { int kr = threadIdx.x >> 3, cb = (threadIdx.x & 7) * 16;
      const float* src = Bb + (size_t)(k0 + kr) * kM + cb0 + cb;
      for (int j = 0; j < 16; j++) Bt[(cb + j)*kSAf + kr] = src[j]; }
    __syncthreads();
    #pragma unroll
    for (int ks = 0; ks < 8; ks++){
      v2f a;
      a.x = As[(wm*16 + lr)*kSAf + ks*4 + k2];
      a.y = As[(wm*16 + lr)*kSAf + ks*4 + k2 + 1];
      #pragma unroll
      for (int jt = 0; jt < 4; jt++){
        int col = wn*64 + jt*16 + lr;
        v2f b;
        b.x = Bt[col*kSAf + ks*4 + k2];
        b.y = Bt[col*kSAf + ks*4 + k2 + 1];
        acc[jt].v = wmma_f32(a, b, acc[jt].v);
      }
    }
    __syncthreads();
  }
  #pragma unroll
  for (int jt = 0; jt < 4; jt++)
    for (int i = 0; i < 8; i++){
      int row = rb0 + wm*16 + i + 8*(lane >> 4);
      int col = cb0 + wn*64 + jt*16 + lr;
      float r = coefAB * acc[jt].f[i];
      if (coefAdd != 0.f) r += coefAdd * Add[(size_t)bh * 65536 + (size_t)row * kM + col];
      if (row == col) r += coefI;
      Cb[(size_t)row * kM + col] = r;
    }
}

// z (f32) -> z^T (bf16) for K-contiguous B-fragments
__global__ void zT_convert(const float* __restrict__ z, bf16_t* __restrict__ zT)
{
  int t = blockIdx.x * 512 + threadIdx.x;
  int bh = t >> 16, j = (t >> 8) & 255, m = t & 255;
  zT[t] = f2bf(z[(size_t)bh * 65536 + (size_t)m * kM + j]);
}

// ================= 7) attn3v = softmax(qland@k^T) @ v  (flash, split over n) =================
__global__ void attn3v_partial(const bf16_t* __restrict__ ql, const bf16_t* __restrict__ k,
                               const bf16_t* __restrict__ v, float* __restrict__ o_part,
                               float* __restrict__ m_part, float* __restrict__ l_part)
{
  int bh = blockIdx.x >> 2, quad = blockIdx.x & 3, split = blockIdx.y;
  int w = threadIdx.x >> 5, lane = threadIdx.x & 31;
  int lr = lane & 15, kh = lane >> 4;
  int rbase = quad * 64 + w * 16;
  __shared__ __align__(16) bf16_t kch[128 * kSK];        // k rows (K=d contiguous)
  __shared__ __align__(16) bf16_t vT[64 * kSV];          // v transposed (K=n contiguous)
  __shared__ __align__(16) bf16_t pl[4][16 * kSV];       // per-wave P staging
  const bf16_t* qbb = ql + (size_t)bh * kM * kDh;
  BFrag a0, a1;
  ldA(a0, qbb, kDh, rbase + lr, 0, kh);
  ldA(a1, qbb, kDh, rbase + lr, 32, kh);
  float mrow[8], lrow[8];
  FAcc o[4];
  #pragma unroll
  for (int i = 0; i < 8; i++){ mrow[i] = -1e30f; lrow[i] = 0.f; }
  #pragma unroll
  for (int dt = 0; dt < 4; dt++) for (int i = 0; i < 8; i++) o[dt].f[i] = 0.f;

  for (int c = 0; c < kN / (128 * kSplit); c++){
    int n0 = split * (kN / kSplit) + c * 128;
    for (int idx = threadIdx.x; idx < 1024; idx += 128){          // 128x64 in 8-elem groups
      int r = idx >> 3, c8 = (idx & 7) * 8;
      *(uint4*)&kch[r*kSK + c8] = *(const uint4*)(k + ((size_t)bh * kN + n0 + r) * kDh + c8);
    }
    for (int idx = threadIdx.x; idx < 1024; idx += 128){
      int r = idx >> 3, c8 = (idx & 7) * 8;
      uint4 val = *(const uint4*)(v + ((size_t)bh * kN + n0 + r) * kDh + c8);
      const bf16_t* sv = (const bf16_t*)&val;
      for (int j = 0; j < 8; j++) vT[(c8 + j)*kSV + r] = sv[j];
    }
    __syncthreads();
    FAcc S[8];
    for (int jt = 0; jt < 8; jt++){
      for (int i = 0; i < 8; i++) S[jt].f[i] = 0.f;
      BFrag b0, b1;
      ldB(b0, kch, kSK, jt*16 + lr, 0, kh);
      ldB(b1, kch, kSK, jt*16 + lr, 32, kh);
      S[jt].v = wmma_bf16(a0.v, b0.v, S[jt].v);
      S[jt].v = wmma_bf16(a1.v, b1.v, S[jt].v);
    }
    #pragma unroll
    for (int i = 0; i < 8; i++){
      float mx = -1e30f;
      for (int jt = 0; jt < 8; jt++) mx = fmaxf(mx, S[jt].f[i]);
      mx = hredMax(mx);
      float nm = fmaxf(mrow[i], mx);
      float corr = __expf(mrow[i] - nm);
      mrow[i] = nm;
      float s = 0.f;
      int prow = i + 8*kh;
      for (int jt = 0; jt < 8; jt++){
        float e = __expf(S[jt].f[i] - nm);
        s += e;
        pl[w][prow*kSV + jt*16 + lr] = f2bf(e);
      }
      s = hredSum(s);
      lrow[i] = lrow[i] * corr + s;
      for (int dt = 0; dt < 4; dt++) o[dt].f[i] *= corr;
    }
    __syncthreads();
    #pragma unroll
    for (int ks = 0; ks < 4; ks++){
      BFrag ap; ldA(ap, pl[w], kSV, lr, ks*32, kh);
      #pragma unroll
      for (int dt = 0; dt < 4; dt++){
        BFrag bv; ldB(bv, vT, kSV, dt*16 + lr, ks*32, kh);
        o[dt].v = wmma_bf16(ap.v, bv.v, o[dt].v);
      }
    }
    __syncthreads();
  }
  size_t pb = ((size_t)split * kBH + bh) * kM;
  #pragma unroll
  for (int i = 0; i < 8; i++){
    int row = rbase + i + 8*kh;
    if (lr == 0){ m_part[pb + row] = mrow[i]; l_part[pb + row] = lrow[i]; }
    for (int dt = 0; dt < 4; dt++)
      o_part[(pb + row) * kDh + dt*16 + lr] = o[dt].f[i];
  }
}

__global__ void attn3v_merge(const float* __restrict__ o_part, const float* __restrict__ m_part,
                             const float* __restrict__ l_part, bf16_t* __restrict__ a3vT)
{
  int row = blockIdx.x & (kM - 1), bh = blockIdx.x >> 8, d = threadIdx.x;
  float gm = -1e30f;
  for (int s = 0; s < kSplit; s++)
    gm = fmaxf(gm, m_part[((size_t)s * kBH + bh) * kM + row]);
  float L = 0.f, acc = 0.f;
  for (int s = 0; s < kSplit; s++){
    size_t pb = ((size_t)s * kBH + bh) * kM + row;
    float sc = __expf(m_part[pb] - gm);
    L += l_part[pb] * sc;
    acc += o_part[pb * kDh + d] * sc;
  }
  a3vT[((size_t)bh * kDh + d) * kM + row] = f2bf(acc / L);    // transposed for B-frags
}

// ================= 8) depthwise conv residual over n =================
__global__ void conv_kernel(const bf16_t* __restrict__ v, const float* __restrict__ rk,
                            float* __restrict__ convb)
{
  int bh = blockIdx.x >> 6, seg = blockIdx.x & 63;
  int n0 = seg * 128, hh = bh & 7;
  __shared__ float vl[160][64];
  __shared__ float wk[33];
  int t = threadIdx.x;
  if (t < 33) wk[t] = rk[hh * 33 + t];
  for (int idx = t; idx < 160 * 64; idx += 256){
    int r = idx >> 6, d = idx & 63;
    int n = n0 - 16 + r;
    float val = 0.f;
    if (n >= 0 && n < kN) val = bf2f(v[((size_t)bh * kN + n) * kDh + d]);
    vl[r][d] = val;
  }
  __syncthreads();
  int d = t & 63, nl0 = (t >> 6) * 32;
  for (int nl = nl0; nl < nl0 + 32; ++nl){
    float s = 0.f;
    #pragma unroll
    for (int kk = 0; kk < 33; ++kk) s += wk[kk] * vl[nl + kk][d];
    convb[((size_t)bh * kN + n0 + nl) * kDh + d] = s;
  }
}

// ================= 9) fused: softmax(q@kland^T) @ Z @ attn3v  + conv =================
__global__ void attn1_fused(const bf16_t* __restrict__ q, const bf16_t* __restrict__ kl,
                            const bf16_t* __restrict__ zT, const bf16_t* __restrict__ a3vT,
                            const float* __restrict__ convb, bf16_t* __restrict__ out_inner)
{
  int bh = blockIdx.x >> 7, seg = blockIdx.x & 127;
  int w = threadIdx.x >> 5, lane = threadIdx.x & 31;
  int lr = lane & 15, kh = lane >> 4;
  int n0 = seg * 64 + w * 16;
  __shared__ __align__(16) bf16_t pl[4][16 * kSP];     // ~33 KB
  const bf16_t* qbb = q + (size_t)bh * kN * kDh;
  const bf16_t* klb = kl + (size_t)bh * kM * kDh;
  BFrag a0, a1;
  ldA(a0, qbb, kDh, n0 + lr, 0, kh);
  ldA(a1, qbb, kDh, n0 + lr, 32, kh);
  FAcc S[16];
  for (int jt = 0; jt < 16; jt++){
    for (int i = 0; i < 8; i++) S[jt].f[i] = 0.f;
    BFrag b0, b1;
    ldB(b0, klb, kDh, jt*16 + lr, 0, kh);
    ldB(b1, klb, kDh, jt*16 + lr, 32, kh);
    S[jt].v = wmma_bf16(a0.v, b0.v, S[jt].v);
    S[jt].v = wmma_bf16(a1.v, b1.v, S[jt].v);
  }
  // full-row softmax (m = 256 fits in registers), write normalized P to LDS
  #pragma unroll
  for (int i = 0; i < 8; i++){
    float mx = -1e30f;
    for (int jt = 0; jt < 16; jt++) mx = fmaxf(mx, S[jt].f[i]);
    mx = hredMax(mx);
    float sum = 0.f;
    for (int jt = 0; jt < 16; jt++){ float e = __expf(S[jt].f[i] - mx); S[jt].f[i] = e; sum += e; }
    sum = hredSum(sum);
    float inv = 1.f / sum;
    int prow = i + 8*kh;
    for (int jt = 0; jt < 16; jt++)
      pl[w][prow*kSP + jt*16 + lr] = f2bf(S[jt].f[i] * inv);
  }
  __syncthreads();
  // B1 = P @ Z  (Z^T bf16 in global, L2-resident)
  BFrag ap[8];
  #pragma unroll
  for (int ks = 0; ks < 8; ks++) ldA(ap[ks], pl[w], kSP, lr, ks*32, kh);
  const bf16_t* ztb = zT + (size_t)bh * kM * kM;
  for (int jt = 0; jt < 16; jt++){
    if (jt + 1 < 16)
      __builtin_prefetch(ztb + (size_t)((jt+1)*16 + lr) * kM + kh*16, 0, 1);
    for (int i = 0; i < 8; i++) S[jt].f[i] = 0.f;      // reuse S registers as B1
    #pragma unroll
    for (int ks = 0; ks < 8; ks++){
      BFrag bz; ldB(bz, ztb, kM, jt*16 + lr, ks*32, kh);
      S[jt].v = wmma_bf16(ap[ks].v, bz.v, S[jt].v);
    }
  }
  __syncthreads();
  for (int jt = 0; jt < 16; jt++)
    for (int i = 0; i < 8; i++)
      pl[w][(i + 8*kh)*kSP + jt*16 + lr] = f2bf(S[jt].f[i]);
  __syncthreads();
  #pragma unroll
  for (int ks = 0; ks < 8; ks++) ldA(ap[ks], pl[w], kSP, lr, ks*32, kh);
  // out = B1 @ attn3v, accumulated on top of the conv residual
  const bf16_t* avb = a3vT + (size_t)bh * kDh * kM;
  const float* cvb = convb + ((size_t)bh * kN + n0) * kDh;
  FAcc O[4];
  for (int dt = 0; dt < 4; dt++){
    __builtin_prefetch(avb + (size_t)(dt*16 + lr) * kM + kh*16, 0, 1);
    for (int i = 0; i < 8; i++)
      O[dt].f[i] = cvb[(size_t)(i + 8*kh) * kDh + dt*16 + lr];
    #pragma unroll
    for (int ks = 0; ks < 8; ks++){
      BFrag bv; ldB(bv, avb, kM, dt*16 + lr, ks*32, kh);
      O[dt].v = wmma_bf16(ap[ks].v, bv.v, O[dt].v);
    }
  }
  int b = bh >> 3, hh = bh & 7;
  for (int dt = 0; dt < 4; dt++)
    for (int i = 0; i < 8; i++){
      int n = n0 + i + 8*kh;
      out_inner[((size_t)b * kN + n) * kDim + hh*kDh + dt*16 + lr] = f2bf(O[dt].f[i]);
    }
}

// ================= 10) out = x + out_inner @ w_out + b_out =================
__global__ void out_proj(const bf16_t* __restrict__ oi, const bf16_t* __restrict__ wT,
                         const float* __restrict__ bo, const float* __restrict__ x,
                         float* __restrict__ out)
{
  int rb0 = blockIdx.x * 64, cb0 = blockIdx.y * 128;
  int w = threadIdx.x >> 5, lane = threadIdx.x & 31;
  int lr = lane & 15, kh = lane >> 4;
  int wm = w >> 1, wn = w & 1;
  __shared__ __align__(16) bf16_t As[64 * kSA];
  __shared__ __align__(16) bf16_t Bt[128 * kSB];
  FAcc acc[4];
  #pragma unroll
  for (int jt = 0; jt < 4; jt++) for (int i = 0; i < 8; i++) acc[jt].f[i] = 0.f;
  for (int k0 = 0; k0 < kDim; k0 += 32){
    { int r = threadIdx.x >> 2, c8 = (threadIdx.x & 3) * 8;
      *(uint4*)&As[r*kSA + c8] = *(const uint4*)(oi + (size_t)(rb0 + r) * kDim + k0 + c8); }
    #pragma unroll
    for (int g = 0; g < 2; g++){
      int gi = threadIdx.x * 2 + g;
      int col = gi >> 2, k8 = (gi & 3) * 8;
      *(uint4*)&Bt[col*kSB + k8] =
          *(const uint4*)(wT + (size_t)(cb0 + col) * kDim + k0 + k8);
    }
    __syncthreads();
    BFrag a; ldA(a, As, kSA, wm*16 + lr, 0, kh);
    #pragma unroll
    for (int jt = 0; jt < 4; jt++){
      BFrag b; ldB(b, Bt, kSB, wn*64 + jt*16 + lr, 0, kh);
      acc[jt].v = wmma_bf16(a.v, b.v, acc[jt].v);
    }
    __syncthreads();
  }
  #pragma unroll
  for (int jt = 0; jt < 4; jt++)
    for (int i = 0; i < 8; i++){
      int row = rb0 + wm*16 + i + 8*kh;
      int col = cb0 + wn*64 + jt*16 + lr;
      out[(size_t)row * kDim + col] =
          acc[jt].f[i] + bo[col] + x[(size_t)row * kDim + col];
    }
}

// ================= host launcher =================
extern "C" void kernel_launch(void* const* d_in, const int* in_sizes, int n_in,
                              void* d_out, int out_size, void* d_ws, size_t ws_size,
                              hipStream_t stream)
{
  (void)in_sizes; (void)n_in; (void)out_size; (void)ws_size;
  const float* x    = (const float*)d_in[0];
  const float* lng  = (const float*)d_in[1];
  const float* lnb  = (const float*)d_in[2];
  const float* wqkv = (const float*)d_in[3];
  const float* wout = (const float*)d_in[4];
  const float* bout = (const float*)d_in[5];
  const float* rker = (const float*)d_in[6];
  float* out = (float*)d_out;

  char* p = (char*)d_ws;
  auto take = [&](size_t bytes) -> char* {
    char* r = p;
    p += (bytes + 255) & ~size_t(255);
    return r;
  };
  bf16_t* h_bf   = (bf16_t*)take((size_t)kNT * kDim * 2);
  bf16_t* wqkvT  = (bf16_t*)take((size_t)kDim * 3 * kDim * 2);
  bf16_t* woutT  = (bf16_t*)take((size_t)kDim * kDim * 2);
  bf16_t* qb     = (bf16_t*)take((size_t)kBH * kN * kDh * 2);
  bf16_t* kb     = (bf16_t*)take((size_t)kBH * kN * kDh * 2);
  bf16_t* vb     = (bf16_t*)take((size_t)kBH * kN * kDh * 2);
  bf16_t* qland  = (bf16_t*)take((size_t)kBH * kM * kDh * 2);
  bf16_t* kland  = (bf16_t*)take((size_t)kBH * kM * kDh * 2);
  float*  attn2  = (float*)take((size_t)kBH * kM * kM * 4);
  float*  z0     = (float*)take((size_t)kBH * kM * kM * 4);
  float*  z1     = (float*)take((size_t)kBH * kM * kM * 4);
  float*  xz     = (float*)take((size_t)kBH * kM * kM * 4);
  float*  w0     = (float*)take((size_t)kBH * kM * kM * 4);
  float*  w1     = (float*)take((size_t)kBH * kM * kM * 4);
  bf16_t* zTb    = (bf16_t*)take((size_t)kBH * kM * kM * 2);
  float*  o_part = (float*)take((size_t)kSplit * kBH * kM * kDh * 4);
  float*  m_part = (float*)take((size_t)kSplit * kBH * kM * 4);
  float*  l_part = (float*)take((size_t)kSplit * kBH * kM * 4);
  bf16_t* a3vT   = (bf16_t*)take((size_t)kBH * kDh * kM * 2);
  float*  convb  = (float*)take((size_t)kBH * kN * kDh * 4);
  bf16_t* oi     = (bf16_t*)take((size_t)kNT * kDim * 2);
  float*  scal   = (float*)take(256);

  convert_wT<<<(kDim * 3 * kDim) / 256, 256, 0, stream>>>(wqkv, wqkvT, kDim, 3 * kDim);
  convert_wT<<<(kDim * kDim) / 256, 256, 0, stream>>>(wout, woutT, kDim, kDim);
  ln_kernel<<<kNT / 8, 256, 0, stream>>>(x, lng, lnb, h_bf);
  qkv_gemm<<<dim3(kNT / 64, (3 * kDim) / 128), 256, 0, stream>>>(h_bf, wqkvT, qb, kb, vb);
  landmark_kernel<<<kBH * kM, kDh, 0, stream>>>(qb, kb, qland, kland);
  sim2_softmax<<<kBH * 2, 256, 0, stream>>>(qland, kland, attn2);
  init_scalar<<<1, 32, 0, stream>>>(scal);
  colsum_max<<<kBH, 256, 0, stream>>>(attn2, scal);
  pinv_init<<<(kBH * kM * kM) / 512, 512, 0, stream>>>(attn2, scal, z0);

  float* zc = z0;
  float* zn = z1;
  for (int it = 0; it < 6; ++it){
    dim3 g(4, 2, kBH);
    gemm256<<<g, 256, 0, stream>>>(attn2, zc, nullptr, xz, 0.f, 0.f, 1.f);   // XZ = X@Z
    gemm256<<<g, 256, 0, stream>>>(xz, xz, xz, w0, 15.f, -7.f, 1.f);         // 15I - 7XZ + XZ@XZ
    gemm256<<<g, 256, 0, stream>>>(xz, w0, nullptr, w1, 13.f, 0.f, -1.f);    // 13I - XZ@W
    gemm256<<<g, 256, 0, stream>>>(zc, w1, nullptr, zn, 0.f, 0.f, 0.25f);    // Z = 0.25 Z@W3
    float* t = zc; zc = zn; zn = t;
  }
  zT_convert<<<(kBH * kM * kM) / 512, 512, 0, stream>>>(zc, zTb);

  attn3v_partial<<<dim3(kBH * 4, kSplit), 128, 0, stream>>>(qland, kb, vb, o_part, m_part, l_part);
  attn3v_merge<<<kBH * kM, kDh, 0, stream>>>(o_part, m_part, l_part, a3vT);
  conv_kernel<<<kBH * 64, 256, 0, stream>>>(vb, rker, convb);
  attn1_fused<<<kBH * 128, 128, 0, stream>>>(qb, kland, zTb, a3vT, convb, oi);
  out_proj<<<dim3(kNT / 64, kDim / 128), 256, 0, stream>>>(oi, woutT, bout, x, out);
}
